// PeriorProbAttention_9010841387423
// MI455X (gfx1250) — matrix operand order repaired and checked by reading, not compiled, not run on hardware
//
#include <hip/hip_runtime.h>
#include <hip/hip_bf16.h>
#include <math.h>

// ---------------------------------------------------------------------------
// ProbSparse (Informer) attention for gfx1250 / MI455X, wave32 + WMMA bf16
// + Tensor Data Mover staging of K/V tiles into LDS.
// Shapes fixed by the reference: B=16, L=4096, H=8, D=64, S=45 samples, u=45.
// ---------------------------------------------------------------------------

typedef __bf16 bf16_t;
typedef __attribute__((ext_vector_type(16))) __bf16 v16bf;
typedef __attribute__((ext_vector_type(8)))  float  v8f;
typedef unsigned int u32x4 __attribute__((ext_vector_type(4)));
typedef int          i32x4 __attribute__((ext_vector_type(4)));
typedef int          i32x8 __attribute__((ext_vector_type(8)));

#define B_    16
#define L_    4096
#define H_    8
#define D_    64
#define BH_   (B_ * H_)
#define HD_   (H_ * D_)
#define SMAX  48            // 45 padded to 3 WMMA tiles of 16
#define CHUNK_ 256          // cumsum chunk rows
#define NCH_  (L_ / CHUNK_) // 16 chunks
#define NEG_INF (-1e30f)

#if defined(__gfx1250__) && __has_builtin(__builtin_amdgcn_tensor_load_to_lds)
#define HAVE_TDM 1
#else
#define HAVE_TDM 0
#endif

// ---- WMMA fragment loaders (contiguous k order, wave32) -------------------
// A-matrix 16x32 bf16: lane group g = lane>>4; element j -> k = g*16 + j
__device__ __forceinline__ v16bf load_a_frag(const bf16_t* p, int lane) {
  int g = (lane >> 4) & 1;
  v16bf f;
#pragma unroll
  for (int j = 0; j < 16; ++j) {
    int k = g * 16 + j;
    f[j] = p[k];
  }
  return f;
}
// B-matrix 32x16 bf16: lane group g holds contiguous K half: k = g*16 + j
__device__ __forceinline__ v16bf load_b_frag(const bf16_t* p, int kstride, int lane) {
  int g = (lane >> 4) & 1;
  v16bf f;
#pragma unroll
  for (int j = 0; j < 16; ++j) {
    int k = g * 16 + j;
    f[j] = p[k * kstride];
  }
  return f;
}
// Same as load_b_frag but source is f32 in LDS (TDM-staged), cvt on read.
__device__ __forceinline__ v16bf load_b_frag_f32(const float* p, int kstride, int lane) {
  int g = (lane >> 4) & 1;
  v16bf f;
#pragma unroll
  for (int j = 0; j < 16; ++j) {
    int k = g * 16 + j;
    f[j] = (bf16_t)p[k * kstride];
  }
  return f;
}

#if HAVE_TDM
// LDS byte offset == low 32 bits of the flat address.
__device__ __forceinline__ unsigned lds_addr_of(const void* p) {
  return (unsigned)(unsigned long long)(size_t)p;
}
// Issue one TDM 2D-tile load: tile_dim0=64 f32 (contiguous), tile_dim1=32 rows,
// row stride = HD_ elements. tdim1 = remaining rows (OOB rows read as zero).
__device__ __forceinline__ void tdm_load_tile_f32(unsigned lds_byte_addr,
                                                  unsigned long long gaddr,
                                                  unsigned tdim1) {
  u32x4 g0;
  g0.x = 1u;                                                  // count=1, user mode
  g0.y = lds_byte_addr;                                       // lds_addr
  g0.z = (unsigned)(gaddr & 0xffffffffull);                   // global_addr[31:0]
  g0.w = (unsigned)((gaddr >> 32) & 0x01ffffffull)            // global_addr[56:32]
         | (2u << 30);                                        // type=2 (image)
  i32x8 g1;
  g1[0] = (2 << 16);                                          // data_size=4B; no mcast
  g1[1] = (int)(64u << 16);                                   // tensor_dim0[15:0]=64
  g1[2] = (int)((tdim1 & 0xffffu) << 16);                     // dim0 hi=0 | dim1[15:0]
  g1[3] = (int)((64u << 16) | ((tdim1 >> 16) & 0xffffu));     // dim1 hi | tile_dim0=64
  g1[4] = 32;                                                 // tile_dim1=32, tile_dim2=0
  g1[5] = (int)HD_;                                           // tensor_dim0_stride=512
  g1[6] = 0;
  g1[7] = 0;
  i32x4 z4 = {0, 0, 0, 0};
#if __clang_major__ >= 23
  i32x8 z8 = {0, 0, 0, 0, 0, 0, 0, 0};
  __builtin_amdgcn_tensor_load_to_lds(g0, g1, z4, z4, z8, 0);
#else
  __builtin_amdgcn_tensor_load_to_lds(g0, g1, z4, z4, 0);
#endif
}
#endif // HAVE_TDM

// ---------------------------------------------------------------------------
// Kernel A: sampling metric M[b,h,q] = max_s(Q.K2^T) - sum_s(Q.K2^T)/L
// grid (BH, L/128), block 256 (8 waves; wave w owns q-tile of 16 rows)
// ---------------------------------------------------------------------------
__global__ void kA_sample_metric(const float* __restrict__ Q,
                                 const float* __restrict__ K,
                                 const int*   __restrict__ idxK,
                                 float* __restrict__ Mws, int S) {
  __shared__ bf16_t k2s[SMAX * 64];      // sampled K rows, squared, bf16
  __shared__ bf16_t qs[8][16 * 64];      // per-wave Q tile
  __shared__ float  st[8][16 * SMAX];    // per-wave score staging

  int bh = blockIdx.x, b = bh / H_, h = bh % H_;
  int tid = threadIdx.x;
  int qblk = blockIdx.y * 128;

  for (int i = tid; i < SMAX * 64; i += 256) {
    int s = i >> 6, d = i & 63;
    float v = 0.f;
    if (s < S) {
      int l = idxK[s];
      float kv = K[((size_t)b * L_ + l) * HD_ + h * D_ + d];
      v = kv * kv;
    }
    k2s[i] = (bf16_t)v;
  }
  for (int i = tid; i < 8 * 16 * 64; i += 256) {
    int w = i >> 10, r = (i >> 6) & 15, d = i & 63;
    int q = qblk + w * 16 + r;
    qs[w][(r << 6) + d] = (bf16_t)Q[((size_t)b * L_ + q) * HD_ + h * D_ + d];
  }
  __syncthreads();

  int w = tid >> 5, lane = tid & 31;
  int g = lane >> 4, col = lane & 15;
  v8f c0 = {}, c1 = {}, c2 = {};
  const bf16_t* qrow = &qs[w][col << 6];
#pragma unroll
  for (int kb = 0; kb < 64; kb += 32) {
    v16bf a  = load_a_frag(qrow + kb, lane);
    v16bf b0 = load_b_frag(&k2s[((0 * 16 + col) << 6) + kb], 1, lane);
    v16bf b1 = load_b_frag(&k2s[((1 * 16 + col) << 6) + kb], 1, lane);
    v16bf b2 = load_b_frag(&k2s[((2 * 16 + col) << 6) + kb], 1, lane);
    c0 = __builtin_amdgcn_wmma_f32_16x16x32_bf16(false, a, false, b0, (short)0, c0, false, false);
    c1 = __builtin_amdgcn_wmma_f32_16x16x32_bf16(false, a, false, b1, (short)0, c1, false, false);
    c2 = __builtin_amdgcn_wmma_f32_16x16x32_bf16(false, a, false, b2, (short)0, c2, false, false);
  }
#pragma unroll
  for (int r = 0; r < 8; ++r) {
    int row = r + 8 * g;
    st[w][row * SMAX + 0 * 16 + col] = c0[r];
    st[w][row * SMAX + 1 * 16 + col] = c1[r];
    st[w][row * SMAX + 2 * 16 + col] = c2[r];
  }
  __syncthreads();
  if (lane < 16) {
    int row = lane;
    float mx = -INFINITY, sm = 0.f;
    for (int s = 0; s < S; ++s) {
      float v = st[w][row * SMAX + s];
      mx = fmaxf(mx, v);
      sm += v;
    }
    Mws[(size_t)bh * L_ + qblk + w * 16 + row] = mx - sm / (float)L_;
  }
}

// ---------------------------------------------------------------------------
// Kernel B: top-U selection per (b,h): U passes of argmax-with-exclusion.
// ---------------------------------------------------------------------------
__global__ void kB_topk(float* __restrict__ Mws, int* __restrict__ Mtop, int U) {
  __shared__ float sv[256];
  __shared__ int   si[256];
  int bh = blockIdx.x, tid = threadIdx.x;
  float* Mp = Mws + (size_t)bh * L_;
  for (int pass = 0; pass < U; ++pass) {
    float bv = -INFINITY; int bi = L_;
    for (int i = tid; i < L_; i += 256) {
      float m = Mp[i];
      if (m > bv) { bv = m; bi = i; }
    }
    sv[tid] = bv; si[tid] = bi;
    __syncthreads();
    for (int s = 128; s > 0; s >>= 1) {
      if (tid < s) {
        if (sv[tid + s] > sv[tid] ||
            (sv[tid + s] == sv[tid] && si[tid + s] < si[tid])) {
          sv[tid] = sv[tid + s]; si[tid] = si[tid + s];
        }
      }
      __syncthreads();
    }
    if (tid == 0) {
      Mtop[bh * SMAX + pass] = si[0];
      Mp[si[0]] = -INFINITY;
    }
    __syncthreads();
  }
}

// ---------------------------------------------------------------------------
// Kernel C: flash attention over selected rows. grid BH, block 128 (4 waves;
// waves 0..2 own q-tiles of 16). K/V chunks staged to LDS by the TDM.
// ---------------------------------------------------------------------------
__global__ void kC_attn(const float* __restrict__ Q, const float* __restrict__ K,
                        const float* __restrict__ V, const int* __restrict__ Mtop,
                        float* __restrict__ ctx, int U) {
  __shared__ bf16_t qs3[3][16 * 64];
  __shared__ __align__(16) float kchf[32 * 64];   // TDM-staged K chunk (f32)
  __shared__ __align__(16) float vchf[32 * 64];   // TDM-staged V chunk (f32)
  __shared__ float  pst[3][16 * 32];
  __shared__ bf16_t pbf[3][16 * 32];
  __shared__ int    rowlen[SMAX];
  __shared__ float  mI[SMAX], lI[SMAX], rsc[SMAX];
  __shared__ int    nChunks;

  int bh = blockIdx.x, b = bh / H_, h = bh % H_;
  int tid = threadIdx.x;
  size_t kvbase = (size_t)b * L_ * HD_ + (size_t)h * D_;   // element offset of row 0

  if (tid < SMAX) {
    rowlen[tid] = (tid < U) ? Mtop[bh * SMAX + tid] : -1;
    mI[tid] = NEG_INF; lI[tid] = 0.f; rsc[tid] = 1.f;
  }
  __syncthreads();
  for (int i = tid; i < 3 * 16 * 64; i += 128) {
    int r = i >> 6, d = i & 63;
    float v = 0.f;
    if (r < U) {
      int qrow = rowlen[r];
      v = Q[((size_t)b * L_ + qrow) * HD_ + h * D_ + d];
    }
    qs3[r >> 4][((r & 15) << 6) + d] = (bf16_t)v;
  }
  if (tid == 0) {
    int mm = 0;
    for (int r = 0; r < U; ++r) mm = max(mm, rowlen[r]);
    nChunks = (mm + 32) / 32;   // keys 0..mm inclusive, 32 per chunk
  }
  __syncthreads();

  int w = tid >> 5, lane = tid & 31;
  int g = lane >> 4, col = lane & 15;
  v16bf aq0 = {}, aq1 = {};
  if (w < 3) {
    aq0 = load_a_frag(&qs3[w][col << 6], lane);
    aq1 = load_a_frag(&qs3[w][(col << 6) + 32], lane);
  }
  v8f acc0 = {}, acc1 = {}, acc2 = {}, acc3 = {};
  const float scale = 0.125f;   // 1/sqrt(64)
  int NC = nChunks;

  for (int c = 0; c < NC; ++c) {
    int kb = c * 32;
    __syncthreads();            // protect kchf/vchf reuse across iterations
#if HAVE_TDM
    if (tid < 32) {             // wave 0 issues both tile DMAs
      unsigned tdim1 = (unsigned)(L_ - kb);
      tdm_load_tile_f32(lds_addr_of(kchf),
                        (unsigned long long)(size_t)(K + kvbase + (size_t)kb * HD_),
                        tdim1);
      tdm_load_tile_f32(lds_addr_of(vchf),
                        (unsigned long long)(size_t)(V + kvbase + (size_t)kb * HD_),
                        tdim1);
      __builtin_amdgcn_s_wait_tensorcnt(0);
    }
#else
    for (int i = tid; i < 32 * 64; i += 128) {
      int r = i >> 6, d = i & 63;
      int pos = kb + r;
      float kv = 0.f, vv = 0.f;
      if (pos < L_) {
        size_t base = kvbase + (size_t)pos * HD_ + d;
        kv = K[base]; vv = V[base];
      }
      kchf[i] = kv; vchf[i] = vv;
    }
#endif
    __syncthreads();

    if (w < 3) {                // scores: Q_tile @ K_chunk^T
      v8f s0 = {}, s1 = {};
#pragma unroll
      for (int kk = 0; kk < 64; kk += 32) {
        v16bf a   = (kk == 0) ? aq0 : aq1;
        v16bf bk0 = load_b_frag_f32(&kchf[(col << 6) + kk], 1, lane);
        v16bf bk1 = load_b_frag_f32(&kchf[((col + 16) << 6) + kk], 1, lane);
        s0 = __builtin_amdgcn_wmma_f32_16x16x32_bf16(false, a, false, bk0, (short)0, s0, false, false);
        s1 = __builtin_amdgcn_wmma_f32_16x16x32_bf16(false, a, false, bk1, (short)0, s1, false, false);
      }
#pragma unroll
      for (int r = 0; r < 8; ++r) {
        int row = r + 8 * g;
        int rl = rowlen[w * 16 + row];
        pst[w][row * 32 + col]      = ((kb + col)      <= rl) ? s0[r] * scale : NEG_INF;
        pst[w][row * 32 + 16 + col] = ((kb + 16 + col) <= rl) ? s1[r] * scale : NEG_INF;
      }
    }
    __syncthreads();

    if (w < 3 && lane < 16) {   // online softmax row update
      int row = lane, grow = w * 16 + row;
      float mold = mI[grow], mx = mold;
      for (int j = 0; j < 32; ++j) mx = fmaxf(mx, pst[w][row * 32 + j]);
      float sc;
      if (mold <= NEG_INF * 0.5f) sc = (mx <= NEG_INF * 0.5f) ? 1.f : 0.f;
      else                        sc = expf(mold - mx);
      float sm = 0.f;
      for (int j = 0; j < 32; ++j) {
        float v = pst[w][row * 32 + j];
        float p = (v <= NEG_INF * 0.5f) ? 0.f : expf(v - mx);
        pbf[w][row * 32 + j] = (bf16_t)p;
        sm += p;
      }
      lI[grow] = lI[grow] * sc + sm;
      mI[grow] = mx;
      rsc[grow] = sc;
    }
    __syncthreads();

    if (w < 3) {                // acc rescale + P_tile @ V_chunk
      float scv[8];
#pragma unroll
      for (int r = 0; r < 8; ++r) scv[r] = rsc[w * 16 + r + 8 * g];
#pragma unroll
      for (int r = 0; r < 8; ++r) {
        acc0[r] *= scv[r]; acc1[r] *= scv[r];
        acc2[r] *= scv[r]; acc3[r] *= scv[r];
      }
      v16bf ap  = load_a_frag(&pbf[w][col * 32], lane);
      v16bf bv0 = load_b_frag_f32(&vchf[col + 0],  64, lane);
      v16bf bv1 = load_b_frag_f32(&vchf[col + 16], 64, lane);
      v16bf bv2 = load_b_frag_f32(&vchf[col + 32], 64, lane);
      v16bf bv3 = load_b_frag_f32(&vchf[col + 48], 64, lane);
      acc0 = __builtin_amdgcn_wmma_f32_16x16x32_bf16(false, ap, false, bv0, (short)0, acc0, false, false);
      acc1 = __builtin_amdgcn_wmma_f32_16x16x32_bf16(false, ap, false, bv1, (short)0, acc1, false, false);
      acc2 = __builtin_amdgcn_wmma_f32_16x16x32_bf16(false, ap, false, bv2, (short)0, acc2, false, false);
      acc3 = __builtin_amdgcn_wmma_f32_16x16x32_bf16(false, ap, false, bv3, (short)0, acc3, false, false);
    }
  }
  __syncthreads();

  if (w < 3) {
#pragma unroll
    for (int r = 0; r < 8; ++r) {
      int row = r + 8 * g, grow = w * 16 + row;
      if (grow < U) {
        float inv = 1.f / lI[grow];
        float* o = &ctx[((size_t)bh * SMAX + grow) * 64];
        o[0 * 16 + col] = acc0[r] * inv;
        o[1 * 16 + col] = acc1[r] * inv;
        o[2 * 16 + col] = acc2[r] * inv;
        o[3 * 16 + col] = acc3[r] * inv;
      }
    }
  }
}

// ---------------------------------------------------------------------------
// Cumsum of V along L (3-kernel chunked scan), output in [B,L,H,D].
// ---------------------------------------------------------------------------
__global__ void kD1_chunksum(const float* __restrict__ V, float* __restrict__ cs) {
  int bh = blockIdx.x, c = blockIdx.y, d = threadIdx.x;
  int b = bh / H_, h = bh % H_;
  float s = 0.f;
  size_t base = ((size_t)b * L_ + (size_t)c * CHUNK_) * HD_ + h * D_ + d;
  for (int i = 0; i < CHUNK_; ++i) s += V[base + (size_t)i * HD_];
  cs[((size_t)bh * NCH_ + c) * D_ + d] = s;
}

__global__ void kD2_prefix(float* __restrict__ cs) {
  int bh = blockIdx.x, d = threadIdx.x;
  float run = 0.f;
  for (int c = 0; c < NCH_; ++c) {
    size_t idx = ((size_t)bh * NCH_ + c) * D_ + d;
    float t = cs[idx];
    cs[idx] = run;
    run += t;
  }
}

__global__ void kD3_scan_out(const float* __restrict__ V, const float* __restrict__ cs,
                             float* __restrict__ out) {
  int bh = blockIdx.x, c = blockIdx.y, d = threadIdx.x;
  int b = bh / H_, h = bh % H_;
  float run = cs[((size_t)bh * NCH_ + c) * D_ + d];
  size_t base = ((size_t)b * L_ + (size_t)c * CHUNK_) * HD_ + h * D_ + d;
  for (int i = 0; i < CHUNK_; ++i) {
    run += V[base + (size_t)i * HD_];
    out[base + (size_t)i * HD_] = run;
  }
}

__global__ void kE_scatter(const int* __restrict__ Mtop, const float* __restrict__ ctx,
                           float* __restrict__ out, int U) {
  int bh = blockIdx.x, d = threadIdx.x;
  int b = bh / H_, h = bh % H_;
  for (int u = 0; u < U; ++u) {
    int row = Mtop[bh * SMAX + u];
    out[((size_t)b * L_ + row) * HD_ + h * D_ + d] =
        ctx[((size_t)bh * SMAX + u) * D_ + d];
  }
}

// ---------------------------------------------------------------------------
extern "C" void kernel_launch(void* const* d_in, const int* in_sizes, int n_in,
                              void* d_out, int out_size, void* d_ws, size_t ws_size,
                              hipStream_t stream) {
  const float* Q    = (const float*)d_in[0];
  const float* K    = (const float*)d_in[1];
  const float* V    = (const float*)d_in[2];
  const int*   idxK = (const int*)d_in[3];
  float* out = (float*)d_out;

  int S = in_sizes[3]; if (S > SMAX) S = SMAX;
  int U = (int)fmin(5.0 * ceil(log((double)L_)), (double)L_);  // = 45
  if (U > SMAX) U = SMAX;

  char* ws = (char*)d_ws;
  float* Mws  = (float*)ws;  ws += (size_t)BH_ * L_ * sizeof(float);        // 2 MB
  int*   Mtop = (int*)ws;    ws += (size_t)BH_ * SMAX * sizeof(int);        // 24 KB
  float* ctx  = (float*)ws;  ws += (size_t)BH_ * SMAX * D_ * sizeof(float); // 1.5 MB
  float* cs   = (float*)ws;                                                 // 0.5 MB

  kA_sample_metric<<<dim3(BH_, L_ / 128), 256, 0, stream>>>(Q, K, idxK, Mws, S);
  kB_topk<<<BH_, 256, 0, stream>>>(Mws, Mtop, U);
  kC_attn<<<BH_, 128, 0, stream>>>(Q, K, V, Mtop, ctx, U);
  kD1_chunksum<<<dim3(BH_, NCH_), D_, 0, stream>>>(V, cs);
  kD2_prefix<<<BH_, D_, 0, stream>>>(cs);
  kD3_scan_out<<<dim3(BH_, NCH_), D_, 0, stream>>>(V, cs, out);
  kE_scatter<<<BH_, D_, 0, stream>>>(Mtop, ctx, out, U);
}
